// HybridBlock_76467597738250
// MI455X (gfx1250) — matrix operand order repaired and checked by reading, not compiled, not run on hardware
//
#include <hip/hip_runtime.h>
#include <hip/hip_bf16.h>
#include <math.h>

typedef __bf16 bf16_t;
typedef __attribute__((ext_vector_type(16))) __bf16 v16bf;
typedef __attribute__((ext_vector_type(8)))  __bf16 v8bf;
typedef __attribute__((ext_vector_type(8)))  float  v8f;

#define NB   2
#define NL   1024
#define ND   1024
#define NF   4096
#define NH   16
#define NHKV 4
#define NHD  64
#define NE   8
#define MTOK (NB*NL)

__device__ __forceinline__ bf16_t f2bf(float f) {
    union { float f; unsigned u; } v; v.f = f;
    unsigned r = v.u + 0x7FFFu + ((v.u >> 16) & 1u);
    unsigned short h = (unsigned short)(r >> 16);
    bf16_t o; __builtin_memcpy(&o, &h, 2); return o;
}
__device__ __forceinline__ float bf2f(bf16_t b) {
    unsigned short h; __builtin_memcpy(&h, &b, 2);
    union { unsigned u; float f; } v; v.u = ((unsigned)h) << 16; return v.f;
}
__device__ __forceinline__ v16bf pack16(const bf16_t* lo, const bf16_t* hi) {
    v8bf a = *(const v8bf*)lo;
    v8bf b = *(const v8bf*)hi;
    v16bf r;
#pragma unroll
    for (int t = 0; t < 8; ++t) { r[t] = a[t]; r[t + 8] = b[t]; }
    return r;
}

// CDNA5 async DMA: global -> LDS, ASYNCcnt-tracked, no VGPR staging.
// INST_OFFSET is added to both the LDS and the global address (ISA 08 §4.4).
__device__ __forceinline__ void async_ld32(const void* g, void* l) {  // 32B per lane
    asm volatile(
        "global_load_async_to_lds_b128 %0, %1, off\n\t"
        "global_load_async_to_lds_b128 %0, %1, off offset:16"
        :: "v"((unsigned)(uintptr_t)l), "v"((unsigned long long)(uintptr_t)g)
        : "memory");
}
__device__ __forceinline__ void async_ld64(const void* g, void* l) {  // 64B per lane
    asm volatile(
        "global_load_async_to_lds_b128 %0, %1, off\n\t"
        "global_load_async_to_lds_b128 %0, %1, off offset:16\n\t"
        "global_load_async_to_lds_b128 %0, %1, off offset:32\n\t"
        "global_load_async_to_lds_b128 %0, %1, off offset:48"
        :: "v"((unsigned)(uintptr_t)l), "v"((unsigned long long)(uintptr_t)g)
        : "memory");
}
#define WAIT_ASYNC_ALL()  asm volatile("s_wait_asynccnt 0x0" ::: "memory")
#define WAIT_ASYNC_KEEP4() asm volatile("s_wait_asynccnt 0x4" ::: "memory")

// ---------------------------------------------------------------- transpose+cast
// src: f32 [K][N] row-major  ->  dst: bf16 [N][K] row-major  (K,N multiples of 32)
__global__ __launch_bounds__(256) void transpose_cast_kernel(
    const float* __restrict__ src, bf16_t* __restrict__ dst, int K, int N) {
    __shared__ float tile[32][33];
    int kb = blockIdx.y * 32, nb = blockIdx.x * 32;
    int tx = threadIdx.x, ty = threadIdx.y;          // 32 x 8
#pragma unroll
    for (int i = 0; i < 32; i += 8)
        tile[ty + i][tx] = src[(long)(kb + ty + i) * N + nb + tx];
    __syncthreads();
#pragma unroll
    for (int i = 0; i < 32; i += 8)
        dst[(long)(nb + ty + i) * K + kb + tx] = f2bf(tile[tx][ty + i]);
}

// ---------------------------------------------------------------- layernorm (fp32 in, bf16 out)
__global__ __launch_bounds__(256) void layernorm_kernel(
    const float* __restrict__ x, const float* __restrict__ g, const float* __restrict__ b,
    bf16_t* __restrict__ out) {
    const int row = blockIdx.x, tid = threadIdx.x;
    const float* xr = x + (long)row * ND;
    __shared__ float red[256];
    float s = 0.f;
    for (int d = tid; d < ND; d += 256) s += xr[d];
    red[tid] = s; __syncthreads();
    for (int o = 128; o > 0; o >>= 1) { if (tid < o) red[tid] += red[tid + o]; __syncthreads(); }
    float mu = red[0] / ND; __syncthreads();
    float v = 0.f;
    for (int d = tid; d < ND; d += 256) { float t = xr[d] - mu; v += t * t; }
    red[tid] = v; __syncthreads();
    for (int o = 128; o > 0; o >>= 1) { if (tid < o) red[tid] += red[tid + o]; __syncthreads(); }
    float inv = rsqrtf(red[0] / ND + 1e-5f);
    for (int d = tid; d < ND; d += 256)
        out[(long)row * ND + d] = f2bf((xr[d] - mu) * inv * g[d] + b[d]);
}

// ---------------------------------------------------------------- generic WMMA GEMM
// C[M,N] = A[M,K] (bf16, row-major) * Bt[N,K]^T (bf16, row-major "B transposed")
// Double-buffered LDS fed by global_load_async_to_lds_b128 (ASYNCcnt pipeline).
// EPI: 0 = store f32; 1 = store f32 acc + aux[m*N+n]; 2 = gelu(acc) -> bf16 store;
//      3 = C[m*N+n] += gate[m*NE+eidx] * acc  (fp32 accumulate-into)
template <int EPI>
__global__ __launch_bounds__(256) void gemm_bf16_kernel(
    const bf16_t* __restrict__ A, const bf16_t* __restrict__ Bt, void* __restrict__ Cv,
    int M, int N, int K, const float* __restrict__ aux, const float* __restrict__ gate, int eidx) {
    constexpr int LDT = 40;                      // 32 + 8 bf16 pad (80B rows, 16B aligned)
    __shared__ bf16_t sA[2][128 * LDT];
    __shared__ bf16_t sB[2][128 * LDT];
    const int tid = threadIdx.x;
    const int lane = tid & 31, wid = tid >> 5;
    const int wm = wid & 3, wn = wid >> 2;       // 4 x 2 wave grid -> wave tile 32(M) x 64(N)
    const int m0 = blockIdx.y * 128, n0 = blockIdx.x * 128;
    const int lr = lane & 15, lh = lane >> 4;
    const int klo = lh * 8, kb16 = lh * 16;
    const int ldrow = tid >> 1, ldcol = (tid & 1) * 16;   // 32B per thread per tile

    v8f c[2][4];
#pragma unroll
    for (int i = 0; i < 2; ++i)
#pragma unroll
        for (int j = 0; j < 4; ++j) c[i][j] = (v8f){};

    const bf16_t* gA = A + (long)(m0 + ldrow) * K + ldcol;
    const bf16_t* gB = Bt + (long)(n0 + ldrow) * K + ldcol;
    const int ldsoff = ldrow * LDT + ldcol;

    const int T = K >> 5;
    async_ld32(gA, &sA[0][ldsoff]);
    async_ld32(gB, &sB[0][ldsoff]);
    int buf = 0;
    for (int t = 0; t < T; ++t) {
        if (t + 1 < T) {
            async_ld32(gA + (long)(t + 1) * 32, &sA[buf ^ 1][ldsoff]);
            async_ld32(gB + (long)(t + 1) * 32, &sB[buf ^ 1][ldsoff]);
            WAIT_ASYNC_KEEP4();        // current tile resident; next 4 DMAs in flight
        } else {
            WAIT_ASYNC_ALL();
        }
        __syncthreads();

        v16bf af[2], bfv[4];
#pragma unroll
        for (int i = 0; i < 2; ++i) {
            const bf16_t* p = &sA[buf][(wm * 32 + i * 16 + lr) * LDT];
            af[i] = pack16(p + klo, p + 16 + klo);            // A 16x32 layout
        }
#pragma unroll
        for (int j = 0; j < 4; ++j) {
            const bf16_t* p = &sB[buf][(wn * 64 + j * 16 + lr) * LDT] + klo;
            bfv[j] = pack16(p, p + 16);                       // B 32x16 layout
        }
#pragma unroll
        for (int i = 0; i < 2; ++i)
#pragma unroll
            for (int j = 0; j < 4; ++j)
                c[i][j] = __builtin_amdgcn_wmma_f32_16x16x32_bf16(
                    false, af[i], false, bfv[j], (short)0, c[i][j], false, false);
        __syncthreads();
        buf ^= 1;
    }

    float* C = (float*)Cv;
    bf16_t* Cb = (bf16_t*)Cv;
#pragma unroll
    for (int i = 0; i < 2; ++i)
#pragma unroll
        for (int j = 0; j < 4; ++j)
#pragma unroll
            for (int r = 0; r < 8; ++r) {
                int m = m0 + wm * 32 + i * 16 + r + 8 * lh;
                int n = n0 + wn * 64 + j * 16 + lr;
                long idx = (long)m * N + n;
                float acc = c[i][j][r];
                if constexpr (EPI == 0) {
                    C[idx] = acc;
                } else if constexpr (EPI == 1) {
                    C[idx] = acc + aux[idx];
                } else if constexpr (EPI == 2) {
                    float gl = 0.5f * acc * (1.f + erff(acc * 0.70710678118f));
                    Cb[idx] = f2bf(gl);
                } else {
                    C[idx] += gate[(long)m * NE + eidx] * acc;
                }
            }
}

// ---------------------------------------------------------------- RoPE + per-head bf16 layouts
// qf [tok][NH*64] -> qbf [b][h][l][64] with rope
__global__ void rope_cast_q_kernel(const float* __restrict__ qf, bf16_t* __restrict__ qbf) {
    long i = (long)blockIdx.x * 256 + threadIdx.x;
    if (i >= (long)MTOK * ND) return;
    int d = i & 63;
    long th = i >> 6;
    int h = th & (NH - 1);
    long tok = th >> 4;
    int lpos = tok & (NL - 1), b = tok >> 10;
    int j = d & 31;
    float invf = __powf(10000.f, -(float)j * (1.f / 32.f));
    float sn, cs; __sincosf((float)lpos * invf, &sn, &cs);
    float xv = qf[i];
    float other = (d < 32) ? -qf[i + 32] : qf[i - 32];
    qbf[(((long)(b * NH + h)) * NL + lpos) * 64 + d] = f2bf(xv * cs + other * sn);
}
// kf [tok][NHKV*64] -> kbf [b][kvh][l][64] with rope
__global__ void rope_cast_k_kernel(const float* __restrict__ kf, bf16_t* __restrict__ kbf) {
    long i = (long)blockIdx.x * 256 + threadIdx.x;
    if (i >= (long)MTOK * NHKV * 64) return;
    int d = i & 63;
    long th = i >> 6;
    int h = th & (NHKV - 1);
    long tok = th >> 2;
    int lpos = tok & (NL - 1), b = tok >> 10;
    int j = d & 31;
    float invf = __powf(10000.f, -(float)j * (1.f / 32.f));
    float sn, cs; __sincosf((float)lpos * invf, &sn, &cs);
    float xv = kf[i];
    float other = (d < 32) ? -kf[i + 32] : kf[i - 32];
    kbf[(((long)(b * NHKV + h)) * NL + lpos) * 64 + d] = f2bf(xv * cs + other * sn);
}
// vf [tok][NHKV*64] -> vt [b][kvh][d][l]  (transposed for the P@V B-operand)
__global__ void cast_transpose_v_kernel(const float* __restrict__ vf, bf16_t* __restrict__ vt) {
    long i = (long)blockIdx.x * 256 + threadIdx.x;
    if (i >= (long)MTOK * NHKV * 64) return;
    int d = i & 63;
    long th = i >> 6;
    int h = th & (NHKV - 1);
    long tok = th >> 2;
    int lpos = tok & (NL - 1), b = tok >> 10;
    vt[(((long)(b * NHKV + h)) * 64 + d) * NL + lpos] = f2bf(vf[i]);
}
__global__ void cast_bf_kernel(const float* __restrict__ src, bf16_t* __restrict__ dst, long n) {
    long i = (long)blockIdx.x * 256 + threadIdx.x;
    if (i < n) dst[i] = f2bf(src[i]);
}

// ---------------------------------------------------------------- flash attention (causal, GQA)
// q [b][h][l][64], k [b][kvh][l][64], vt [b][kvh][64][l] -> ctx f32 [tok][NH*64]
__global__ __launch_bounds__(128) void flash_attn_kernel(
    const bf16_t* __restrict__ q, const bf16_t* __restrict__ k,
    const bf16_t* __restrict__ vt, float* __restrict__ ctx) {
    constexpr int LDK = 72;                       // 64 + 8 pad (144B rows)
    __shared__ bf16_t sK[64 * LDK];               // [key][dim]
    __shared__ bf16_t sV[64 * LDK];               // [dim][key]
    __shared__ bf16_t sP[4 * 16 * LDK];           // per-wave P tile [row][key]
    const int b = blockIdx.z, h = blockIdx.y, qt = blockIdx.x;
    const int kvh = h >> 2;
    const int tid = threadIdx.x, lane = tid & 31, w = tid >> 5;
    const int lr = lane & 15, lh = lane >> 4;
    const int klo = lh * 8, kb16 = lh * 16;
    const int qbase = qt * 64;
    const int qrow = qbase + w * 16 + lr;

    // Q A-fragments (resident for the whole block)
    const bf16_t* qp = q + (((long)(b * NH + h)) * NL + qrow) * 64;
    v16bf qa0 = pack16(qp + klo, qp + 16 + klo);          // hd 0..31
    v16bf qa1 = pack16(qp + 32 + klo, qp + 48 + klo);     // hd 32..63

    v8f o[4];
#pragma unroll
    for (int j = 0; j < 4; ++j) o[j] = (v8f){};
    float mrow[8], lsum[8];
#pragma unroll
    for (int r = 0; r < 8; ++r) { mrow[r] = -3.0e38f; lsum[r] = 0.f; }

    const int ldrow = tid >> 1, ldch = (tid & 1) * 32;    // 64B per thread per tile
    const bf16_t* kg = k + (((long)(b * NHKV + kvh)) * NL + ldrow) * 64 + ldch;
    const bf16_t* vg = vt + (((long)(b * NHKV + kvh)) * 64 + ldrow) * NL + ldch;

    bf16_t* pbase = &sP[w * 16 * LDK];
    const int nkt = qt + 1;
    for (int kt = 0; kt < nkt; ++kt) {
        // async DMA: K/V 64x64 bf16 tiles straight into LDS (no VGPR staging)
        async_ld64(kg + (long)kt * 64 * 64, &sK[ldrow * LDK + ldch]);
        async_ld64(vg + (long)kt * 64, &sV[ldrow * LDK + ldch]);
        WAIT_ASYNC_ALL();
        __syncthreads();

        // S = Q @ K^T   (4 key sub-tiles of 16, hd contraction 64 = 2 chained WMMAs)
        v8f s[4];
#pragma unroll
        for (int j = 0; j < 4; ++j) {
            const bf16_t* kp = &sK[(j * 16 + lr) * LDK];
            v16bf b0 = pack16(kp + klo, kp + 16 + klo);
            v16bf b1 = pack16(kp + 32 + klo, kp + 48 + klo);
            v8f acc = (v8f){};
            acc = __builtin_amdgcn_wmma_f32_16x16x32_bf16(false, qa0, false, b0, (short)0, acc, false, false);
            acc = __builtin_amdgcn_wmma_f32_16x16x32_bf16(false, qa1, false, b1, (short)0, acc, false, false);
            s[j] = acc;
        }

        // online softmax per row (row = r + 8*lh for this lane's C fragments)
#pragma unroll
        for (int r = 0; r < 8; ++r) {
            int mq = qbase + w * 16 + r + 8 * lh;
            float localmax = -3.0e38f;
#pragma unroll
            for (int j = 0; j < 4; ++j) {
                int key = kt * 64 + j * 16 + lr;
                float sv = s[j][r] * 0.125f;
                if (key > mq) sv = -3.0e38f;
                s[j][r] = sv;
                localmax = fmaxf(localmax, sv);
            }
            for (int off = 1; off < 16; off <<= 1)
                localmax = fmaxf(localmax, __shfl_xor(localmax, off, 32));
            float mnew = fmaxf(mrow[r], localmax);
            float corr = __expf(mrow[r] - mnew);
            mrow[r] = mnew;
            float psum = 0.f;
#pragma unroll
            for (int j = 0; j < 4; ++j) {
                float p = __expf(s[j][r] - mnew);
                psum += p;
                pbase[(r + 8 * lh) * LDK + j * 16 + lr] = f2bf(p);
            }
            for (int off = 1; off < 16; off <<= 1)
                psum += __shfl_xor(psum, off, 32);
            lsum[r] = lsum[r] * corr + psum;
#pragma unroll
            for (int j = 0; j < 4; ++j) o[j][r] *= corr;
        }

        // P A-fragments from per-wave LDS (same-wave in-order LDS; no barrier needed)
        const bf16_t* pp = pbase + lr * LDK;
        v16bf pa0 = pack16(pp + klo, pp + 16 + klo);           // keys 0..31
        v16bf pa1 = pack16(pp + 32 + klo, pp + 48 + klo);      // keys 32..63

        // O += P @ V   (4 dim sub-tiles, key contraction 64 = 2 chained WMMAs)
#pragma unroll
        for (int j = 0; j < 4; ++j) {
            const bf16_t* vp = &sV[(j * 16 + lr) * LDK];
            v16bf vb0 = pack16(vp + klo, vp + 16 + klo);
            v16bf vb1 = pack16(vp + 32 + klo, vp + 48 + klo);
            o[j] = __builtin_amdgcn_wmma_f32_16x16x32_bf16(false, pa0, false, vb0, (short)0, o[j], false, false);
            o[j] = __builtin_amdgcn_wmma_f32_16x16x32_bf16(false, pa1, false, vb1, (short)0, o[j], false, false);
        }
        __syncthreads();
    }

    // normalize + write ctx [tok][h*64 + d]
#pragma unroll
    for (int r = 0; r < 8; ++r) {
        float inv = 1.f / lsum[r];
        int mq = qbase + w * 16 + r + 8 * lh;
        long tok = (long)b * NL + mq;
#pragma unroll
        for (int j = 0; j < 4; ++j)
            ctx[tok * ND + h * 64 + j * 16 + lr] = o[j][r] * inv;
    }
}

// ---------------------------------------------------------------- MoE router: top-2 + counts
__global__ __launch_bounds__(256) void router_kernel(
    const bf16_t* __restrict__ h2, const float* __restrict__ wg,
    float* __restrict__ comb, float* __restrict__ counts) {
    const int tok = blockIdx.x;
    const int e = threadIdx.x >> 5, lane = threadIdx.x & 31;
    const bf16_t* hp = h2 + (long)tok * ND;
    float s = 0.f;
    for (int d = lane; d < ND; d += 32) s += bf2f(hp[d]) * wg[(long)d * NE + e];
    for (int off = 16; off > 0; off >>= 1) s += __shfl_xor(s, off, 32);
    __shared__ float logits[NE];
    if (lane == 0) logits[e] = s;
    __syncthreads();
    if (threadIdx.x == 0) {
        int i0 = 0, i1 = 0; float b0 = -1e30f, b1 = -1e30f;
        for (int t = 0; t < NE; ++t) {
            float v = logits[t];
            if (v > b0) { b1 = b0; i1 = i0; b0 = v; i0 = t; }
            else if (v > b1) { b1 = v; i1 = t; }
        }
        float e0 = 1.f, e1 = __expf(b1 - b0), den = e0 + e1;
        float* cr = comb + (long)tok * NE;
        for (int t = 0; t < NE; ++t) cr[t] = 0.f;
        cr[i0] = e0 / den; cr[i1] = e1 / den;
        atomicAdd(&counts[i0], 1.f);
        atomicAdd(&counts[i1], 1.f);
    }
}

__global__ void lb_loss_kernel(const float* __restrict__ counts, float* __restrict__ out) {
    if (threadIdx.x == 0 && blockIdx.x == 0) {
        float tot = 0.f;
        for (int e = 0; e < NE; ++e) tot += counts[e];
        float acc = 0.f;
        for (int e = 0; e < NE; ++e) { float d = counts[e] / tot - 1.f / NE; acc += d * d; }
        out[0] = acc / NE;
    }
}

// ---------------------------------------------------------------- host orchestration
extern "C" void kernel_launch(void* const* d_in, const int* in_sizes, int n_in,
                              void* d_out, int out_size, void* d_ws, size_t ws_size,
                              hipStream_t stream) {
    const float* x    = (const float*)d_in[0];
    const float* wq   = (const float*)d_in[1];
    const float* wk   = (const float*)d_in[2];
    const float* wv   = (const float*)d_in[3];
    const float* wo   = (const float*)d_in[4];
    const float* wg   = (const float*)d_in[5];
    const float* w1   = (const float*)d_in[6];
    const float* w2   = (const float*)d_in[7];
    const float* ln1g = (const float*)d_in[8];
    const float* ln1b = (const float*)d_in[9];
    const float* ln2g = (const float*)d_in[10];
    const float* ln2b = (const float*)d_in[11];
    float* out_x  = (float*)d_out;                       // [B,L,D] f32
    float* out_lb = out_x + (long)MTOK * ND;             // scalar

    char* base = (char*)d_ws; size_t off = 0;
    auto alloc = [&](size_t bytes) -> char* {
        char* p = base + off; off = (off + bytes + 255) & ~(size_t)255; return p;
    };
    bf16_t* wqt  = (bf16_t*)alloc((size_t)ND * ND * 2);
    bf16_t* wkt  = (bf16_t*)alloc((size_t)ND * NHKV * 64 * 2);
    bf16_t* wvt  = (bf16_t*)alloc((size_t)ND * NHKV * 64 * 2);
    bf16_t* wot  = (bf16_t*)alloc((size_t)ND * ND * 2);
    bf16_t* w1t  = (bf16_t*)alloc((size_t)NE * ND * NF * 2);
    bf16_t* w2t  = (bf16_t*)alloc((size_t)NE * NF * ND * 2);
    bf16_t* h1   = (bf16_t*)alloc((size_t)MTOK * ND * 2);
    bf16_t* h2   = (bf16_t*)alloc((size_t)MTOK * ND * 2);
    bf16_t* qbf  = (bf16_t*)alloc((size_t)MTOK * ND * 2);
    bf16_t* kbf  = (bf16_t*)alloc((size_t)MTOK * NHKV * 64 * 2);
    bf16_t* vtbf = (bf16_t*)alloc((size_t)MTOK * NHKV * 64 * 2);
    bf16_t* ctxb = (bf16_t*)alloc((size_t)MTOK * ND * 2);
    bf16_t* hid  = (bf16_t*)alloc((size_t)MTOK * NF * 2);
    float*  qf   = (float*)alloc((size_t)MTOK * ND * 4);
    float*  kf   = (float*)alloc((size_t)MTOK * NHKV * 64 * 4);
    float*  vf   = (float*)alloc((size_t)MTOK * NHKV * 64 * 4);
    float*  ctxf = (float*)alloc((size_t)MTOK * ND * 4);
    float*  x1   = (float*)alloc((size_t)MTOK * ND * 4);
    float*  comb = (float*)alloc((size_t)MTOK * NE * 4);
    float*  cnts = (float*)alloc(64);

    dim3 tb(32, 8);
    // weight casts (transposed -> Bt layout [N][K])
    transpose_cast_kernel<<<dim3(ND / 32, ND / 32), tb, 0, stream>>>(wq, wqt, ND, ND);
    transpose_cast_kernel<<<dim3((NHKV * 64) / 32, ND / 32), tb, 0, stream>>>(wk, wkt, ND, NHKV * 64);
    transpose_cast_kernel<<<dim3((NHKV * 64) / 32, ND / 32), tb, 0, stream>>>(wv, wvt, ND, NHKV * 64);
    transpose_cast_kernel<<<dim3(ND / 32, ND / 32), tb, 0, stream>>>(wo, wot, ND, ND);
    for (int e = 0; e < NE; ++e) {
        transpose_cast_kernel<<<dim3(NF / 32, ND / 32), tb, 0, stream>>>(
            w1 + (size_t)e * ND * NF, w1t + (size_t)e * NF * ND, ND, NF);
        transpose_cast_kernel<<<dim3(ND / 32, NF / 32), tb, 0, stream>>>(
            w2 + (size_t)e * NF * ND, w2t + (size_t)e * ND * NF, NF, ND);
    }

    // LN1 -> h1
    layernorm_kernel<<<MTOK, 256, 0, stream>>>(x, ln1g, ln1b, h1);

    // Q/K/V projections (f32 out for RoPE)
    gemm_bf16_kernel<0><<<dim3(ND / 128, MTOK / 128), 256, 0, stream>>>(
        h1, wqt, qf, MTOK, ND, ND, nullptr, nullptr, 0);
    gemm_bf16_kernel<0><<<dim3((NHKV * 64) / 128, MTOK / 128), 256, 0, stream>>>(
        h1, wkt, kf, MTOK, NHKV * 64, ND, nullptr, nullptr, 0);
    gemm_bf16_kernel<0><<<dim3((NHKV * 64) / 128, MTOK / 128), 256, 0, stream>>>(
        h1, wvt, vf, MTOK, NHKV * 64, ND, nullptr, nullptr, 0);

    // RoPE + per-head bf16 layouts
    rope_cast_q_kernel<<<(int)(((long)MTOK * ND + 255) / 256), 256, 0, stream>>>(qf, qbf);
    rope_cast_k_kernel<<<(int)(((long)MTOK * NHKV * 64 + 255) / 256), 256, 0, stream>>>(kf, kbf);
    cast_transpose_v_kernel<<<(int)(((long)MTOK * NHKV * 64 + 255) / 256), 256, 0, stream>>>(vf, vtbf);

    // flash attention -> ctx f32
    flash_attn_kernel<<<dim3(NL / 64, NH, NB), 128, 0, stream>>>(qbf, kbf, vtbf, ctxf);

    // out-projection + residual: x1 = x + ctx @ wo
    cast_bf_kernel<<<(int)(((long)MTOK * ND + 255) / 256), 256, 0, stream>>>(ctxf, ctxb, (long)MTOK * ND);
    gemm_bf16_kernel<1><<<dim3(ND / 128, MTOK / 128), 256, 0, stream>>>(
        ctxb, wot, x1, MTOK, ND, ND, x, nullptr, 0);

    // LN2 -> h2
    layernorm_kernel<<<MTOK, 256, 0, stream>>>(x1, ln2g, ln2b, h2);

    // router (counts zeroed first)
    hipMemsetAsync(cnts, 0, 64, stream);
    router_kernel<<<MTOK, 256, 0, stream>>>(h2, wg, comb, cnts);

    // init output with residual, then accumulate gated expert outputs
    hipMemcpyAsync(out_x, x1, (size_t)MTOK * ND * 4, hipMemcpyDeviceToDevice, stream);
    for (int e = 0; e < NE; ++e) {
        gemm_bf16_kernel<2><<<dim3(NF / 128, MTOK / 128), 256, 0, stream>>>(
            h2, w1t + (size_t)e * NF * ND, hid, MTOK, NF, ND, nullptr, nullptr, 0);
        gemm_bf16_kernel<3><<<dim3(ND / 128, MTOK / 128), 256, 0, stream>>>(
            hid, w2t + (size_t)e * ND * NF, out_x, MTOK, ND, NF, nullptr, comb, e);
    }

    lb_loss_kernel<<<1, 32, 0, stream>>>(cnts, out_lb);
}
